// CommNet_9431748182144
// MI455X (gfx1250) — matrix operand order repaired and checked by reading, not compiled, not run on hardware
//
#include <hip/hip_runtime.h>
#include <hip/hip_bf16.h>

#define ROWS   131072
#define HID    256
#define OBS    128
#define NACT   32
#define LS     264   // LDS row stride (halfs/floats): 16B-aligned, bank-staggered

typedef __attribute__((ext_vector_type(16))) __bf16 v16bf;
typedef __attribute__((ext_vector_type(8)))  __bf16 v8bf;
typedef __attribute__((ext_vector_type(8)))  float  v8f;

#define WMMA_BF16(a,b,c) \
  __builtin_amdgcn_wmma_f32_16x16x32_bf16(false,(a),false,(b),(short)0,(c),false,false)

// Fast sigmoid: v_exp_f32 + v_rcp_f32 (no IEEE-division expansion).
// exp(-x) -> inf for very negative x; rcp(inf)=0, so saturation is correct.
static __device__ __forceinline__ float sigf(float x) {
  return __builtin_amdgcn_rcpf(1.0f + __expf(-x));
}

// Fast tanh: (e^{2x}-1)/(e^{2x}+1) with clamp so e^{2x} never overflows.
static __device__ __forceinline__ float tanh_fast(float x) {
  float xc = fminf(fmaxf(x, -15.0f), 15.0f);
  float t  = __expf(2.0f * xc);
  return (t - 1.0f) * __builtin_amdgcn_rcpf(t + 1.0f);
}

// A fragment (16x32 bf16, M=row per lane) from row-major LDS [16][LS] bf16.
// Lane l: row m=l&15; lanes 0-15 take K 0..15, lanes 16-31 K 16..31 (matches load_b).
static __device__ __forceinline__ v16bf load_a(const __bf16* base, int lane, int kt) {
  const int m = lane & 15, sel = (lane >> 4) & 1;
  const __bf16* p = base + m * LS + kt * 32 + sel * 16;
  v16bf a;
  ((v8bf*)&a)[0] = *(const v8bf*)(p);        // K = kt*32+sel*16 .. +7  (VGPR 0-3)
  ((v8bf*)&a)[1] = *(const v8bf*)(p + 8);    // K = +8 .. +15           (VGPR 4-7)
  return a;
}

// B fragment (32x16 bf16, N=col per lane) from row-major weights W[out][K]:
// lane l: n=l&15, 16 consecutive K halfs starting at sel*16 -> one 32B load.
static __device__ __forceinline__ v16bf load_b(const __bf16* w, int lane, int n0,
                                               int kt, int kstride) {
  const int n = lane & 15, sel = (lane >> 4) & 1;
  return *(const v16bf*)(w + (size_t)(n0 + n) * kstride + kt * 32 + sel * 16);
}

// One torch-style GRU cell for this wave's 16 rows.
// lx: input activation (bf16 A-source), lh: h snapshot (bf16 A-source),
// hf: fp32 h state [16][LS] updated in place.
static __device__ __forceinline__ void gru_cell(const __bf16* lx, const __bf16* lh,
                                                float* hf,
                                                const __bf16* wih, const __bf16* whh,
                                                const float* bih, const float* bhh,
                                                int lane) {
  const int n = lane & 15, sel = (lane >> 4) & 1;
  for (int j = 0; j < 16; ++j) {           // output column tile (16 cols of H)
    v8f ir = {0,0,0,0,0,0,0,0}, iz = ir, inn = ir, hr = ir, hz = ir, hn = ir;
    #pragma unroll
    for (int kt = 0; kt < 8; ++kt) {       // K = 256 in 32-chunks
      v16bf ax = load_a(lx, lane, kt);
      v16bf ah = load_a(lh, lane, kt);
      ir  = WMMA_BF16(ax, load_b(wih, lane,           j*16, kt, HID), ir);
      iz  = WMMA_BF16(ax, load_b(wih, lane,   HID  +  j*16, kt, HID), iz);
      inn = WMMA_BF16(ax, load_b(wih, lane, 2*HID  +  j*16, kt, HID), inn);
      hr  = WMMA_BF16(ah, load_b(whh, lane,           j*16, kt, HID), hr);
      hz  = WMMA_BF16(ah, load_b(whh, lane,   HID  +  j*16, kt, HID), hz);
      hn  = WMMA_BF16(ah, load_b(whh, lane, 2*HID  +  j*16, kt, HID), hn);
    }
    const int c = j * 16 + n;
    const float bir = bih[c], biz = bih[HID + c], bin = bih[2*HID + c];
    const float bhr = bhh[c], bhz = bhh[HID + c], bhn = bhh[2*HID + c];
    #pragma unroll
    for (int v = 0; v < 8; ++v) {          // D element v -> row m = sel*8+v
      const int m = sel * 8 + v;
      float r  = sigf(ir[v] + bir + hr[v] + bhr);
      float z  = sigf(iz[v] + biz + hz[v] + bhz);
      float nn = tanh_fast(inn[v] + bin + r * (hn[v] + bhn));
      float ho = hf[m * LS + c];
      hf[m * LS + c] = (1.0f - z) * nn + z * ho;
    }
  }
}

// Comm step: c_i = (sum_j h_j - h_i)/16 over this wave's 16-row agent group.
// Writes bf16 c -> a0 (if write_c) and bf16 h snapshot -> a1.
static __device__ __forceinline__ void comm_step(const float* hf, __bf16* a0, __bf16* a1,
                                                 int lane, bool write_c) {
  const int n = lane & 15, sel = (lane >> 4) & 1;
  for (int j = 0; j < 16; ++j) {
    const int c = j * 16 + n;
    float e[8], s = 0.0f;
    #pragma unroll
    for (int v = 0; v < 8; ++v) { e[v] = hf[(sel*8 + v) * LS + c]; s += e[v]; }
    s += __shfl_xor(s, 16, 32);            // combine the two M-halves (lanes l, l^16)
    #pragma unroll
    for (int v = 0; v < 8; ++v) {
      const int m = sel * 8 + v;
      a1[m * LS + c] = (__bf16)e[v];
      if (write_c) a0[m * LS + c] = (__bf16)((s - e[v]) * (1.0f / 16.0f));
    }
  }
}

__global__ void convert_weights(const float* __restrict__ encw, const float* __restrict__ wih,
                                const float* __restrict__ whh,  const float* __restrict__ decw,
                                __bf16* __restrict__ ws) {
  const int n_enc = HID * OBS, n_ih = 3 * HID * HID, n_hh = 3 * HID * HID, n_dec = NACT * HID;
  const int total = n_enc + n_ih + n_hh + n_dec;
  int i = blockIdx.x * blockDim.x + threadIdx.x;
  if (i >= total) return;
  float v;
  if (i < n_enc)                    v = encw[i];
  else if (i < n_enc + n_ih)        v = wih[i - n_enc];
  else if (i < n_enc + n_ih + n_hh) v = whh[i - n_enc - n_ih];
  else                              v = decw[i - n_enc - n_ih - n_hh];
  ws[i] = (__bf16)v;
}

__global__ void commnet_fused(const float* __restrict__ obs,  const float* __restrict__ h0,
                              const float* __restrict__ encb, const float* __restrict__ bih,
                              const float* __restrict__ bhh,  const float* __restrict__ decb,
                              const __bf16* __restrict__ encw, const __bf16* __restrict__ wih,
                              const __bf16* __restrict__ whh,  const __bf16* __restrict__ decw,
                              float* __restrict__ out_w, float* __restrict__ out_h) {
  extern __shared__ char smem[];
  const int lane = threadIdx.x & 31;
  const int wid  = threadIdx.x >> 5;           // 8 waves; each owns one 16-row agent group
  const int row0 = (blockIdx.x * 8 + wid) * 16;
  const int n = lane & 15, sel = (lane >> 4) & 1;

  __bf16* A0 = (__bf16*)smem + (size_t)wid * (16 * LS);          // activation A-source
  __bf16* A1 = (__bf16*)smem + (size_t)(8 + wid) * (16 * LS);    // h / obs A-source
  float*  HF = (float*)((__bf16*)smem + (size_t)16 * (16 * LS)) + (size_t)wid * (16 * LS);

  // ---- stage obs (fp32 -> bf16) into A1 ----
  for (int idx = lane; idx < 16 * OBS; idx += 32) {
    int r = idx >> 7, cc = idx & (OBS - 1);
    A1[r * LS + cc] = (__bf16)obs[(size_t)(row0 + r) * OBS + cc];
  }
  asm volatile("s_wait_dscnt 0x0" ::: "memory");

  // ---- encoder: x = sigmoid(obs @ enc_w^T + enc_b) -> A0 ----
  for (int j = 0; j < 16; ++j) {
    v8f acc = {0,0,0,0,0,0,0,0};
    #pragma unroll
    for (int kt = 0; kt < 4; ++kt)
      acc = WMMA_BF16(load_a(A1, lane, kt), load_b(encw, lane, j * 16, kt, OBS), acc);
    const int c = j * 16 + n;
    const float b = encb[c];
    #pragma unroll
    for (int v = 0; v < 8; ++v)
      A0[(sel * 8 + v) * LS + c] = (__bf16)sigf(acc[v] + b);
  }
  asm volatile("s_wait_dscnt 0x0" ::: "memory");

  // ---- stage h0 -> A1 (bf16) and HF (fp32) ----
  for (int idx = lane; idx < 16 * HID; idx += 32) {
    int r = idx >> 8, cc = idx & (HID - 1);
    float hv = h0[(size_t)(row0 + r) * HID + cc];
    A1[r * LS + cc] = (__bf16)hv;
    HF[r * LS + cc] = hv;
  }
  asm volatile("s_wait_dscnt 0x0" ::: "memory");

  // ---- GRU 1 + two (comm -> GRU) steps ----
  gru_cell(A0, A1, HF, wih, whh, bih, bhh, lane);
  asm volatile("s_wait_dscnt 0x0" ::: "memory");
  for (int step = 0; step < 2; ++step) {
    comm_step(HF, A0, A1, lane, true);
    asm volatile("s_wait_dscnt 0x0" ::: "memory");
    gru_cell(A0, A1, HF, wih, whh, bih, bhh, lane);
    asm volatile("s_wait_dscnt 0x0" ::: "memory");
  }

  // ---- snapshot final h (bf16) into A1 for the decoder ----
  comm_step(HF, A0, A1, lane, false);
  asm volatile("s_wait_dscnt 0x0" ::: "memory");

  // ---- decoder: weights = h @ dec_w^T + dec_b ----
  for (int j = 0; j < 2; ++j) {
    v8f acc = {0,0,0,0,0,0,0,0};
    #pragma unroll
    for (int kt = 0; kt < 8; ++kt)
      acc = WMMA_BF16(load_a(A1, lane, kt), load_b(decw, lane, j * 16, kt, HID), acc);
    const int c = j * 16 + n;
    const float b = decb[c];
    #pragma unroll
    for (int v = 0; v < 8; ++v)
      out_w[(size_t)(row0 + sel * 8 + v) * NACT + c] = acc[v] + b;
  }

  // ---- store final h (fp32) ----
  for (int idx = lane; idx < 16 * HID; idx += 32) {
    int r = idx >> 8, cc = idx & (HID - 1);
    out_h[(size_t)(row0 + r) * HID + cc] = HF[r * LS + cc];
  }
}

extern "C" void kernel_launch(void* const* d_in, const int* in_sizes, int n_in,
                              void* d_out, int out_size, void* d_ws, size_t ws_size,
                              hipStream_t stream) {
  const float* obs  = (const float*)d_in[0];
  const float* h0   = (const float*)d_in[1];
  const float* encw = (const float*)d_in[2];
  const float* encb = (const float*)d_in[3];
  const float* wih  = (const float*)d_in[4];
  const float* whh  = (const float*)d_in[5];
  const float* bih  = (const float*)d_in[6];
  const float* bhh  = (const float*)d_in[7];
  const float* decw = (const float*)d_in[8];
  const float* decb = (const float*)d_in[9];

  const int n_enc = HID * OBS, n_ih = 3 * HID * HID, n_hh = 3 * HID * HID, n_dec = NACT * HID;
  const int total = n_enc + n_ih + n_hh + n_dec;

  __bf16* ws     = (__bf16*)d_ws;
  __bf16* enc_bf = ws;
  __bf16* wih_bf = enc_bf + n_enc;
  __bf16* whh_bf = wih_bf + n_ih;
  __bf16* dec_bf = whh_bf + n_hh;

  convert_weights<<<(total + 255) / 256, 256, 0, stream>>>(encw, wih, whh, decw, ws);

  float* out_w = (float*)d_out;
  float* out_h = out_w + (size_t)ROWS * NACT;

  // LDS: 2 bf16 buffers + 1 fp32 buffer, each 16*LS elems, per wave, 8 waves.
  size_t lds = (size_t)8 * 16 * LS * (2 + 2 + 4);  // = 270336 B
  commnet_fused<<<ROWS / 128, 256, lds, stream>>>(obs, h0, encb, bih, bhh, decb,
                                                  enc_bf, wih_bf, whh_bf, dec_bf,
                                                  out_w, out_h);
}
